// DotProductAttention_90666759618581
// MI455X (gfx1250) — matrix operand order repaired and checked by reading, not compiled, not run on hardware
//
#include <hip/hip_runtime.h>

typedef __attribute__((ext_vector_type(16))) __bf16 v16bf;
typedef __attribute__((ext_vector_type(8)))  __bf16 v8bf;
typedef __attribute__((ext_vector_type(8)))  float  v8f;

#define BATCH    4
#define HEADS    16
#define SEQ      2048
#define DDIM     64
#define BM       64
#define BN       64
#define NBLK     (SEQ / BN)
#define NTHREADS 128
#define KST      72      // bf16 LDS row stride (16B-aligned, padded)
#define SST      68      // fp32 LDS row stride

// --- CDNA5 async global->LDS copy path (ASYNCcnt-tracked) -------------------
__device__ __forceinline__ unsigned lds_off(const void* p) {
    // low 32 bits of a generic shared-aperture address == LDS byte offset
    return (unsigned)(unsigned long long)p;
}
__device__ __forceinline__ void async_copy_b128(unsigned ldsByteOff, const void* gptr) {
    asm volatile("global_load_async_to_lds_b128 %0, %1, off"
                 :: "v"(ldsByteOff), "v"(gptr) : "memory");
}
__device__ __forceinline__ void wait_async0() {
    asm volatile("s_wait_asynccnt 0x0" ::: "memory");
}

// A-matrix 16x32 bf16 fragment from row-major LDS tile (row m, contiguous K).
// ISA layout: lanes 0-15 hold K {0..7,16..23}, lanes 16-31 hold K {8..15,24..31}.
__device__ __forceinline__ v16bf load_a_frag(const __bf16* base, int lane) {
    int row = lane & 15;
    int hi  = lane >> 4;
    const __bf16* p = base + row * KST + hi * 8;
    v8bf lo8 = *(const v8bf*)(p);
    v8bf hi8 = *(const v8bf*)(p + 16);
    return __builtin_shufflevector(lo8, hi8, 0,1,2,3,4,5,6,7,8,9,10,11,12,13,14,15);
}

// B-matrix 32x16 bf16 fragment; source tile stored with B-columns as LDS rows.
// ISA layout: lanes 0-15 hold K {0..7,16..23}, lanes 16-31 hold K {8..15,24..31}.
__device__ __forceinline__ v16bf load_b_frag(const __bf16* base, int lane) {
    int col = lane & 15;
    int hi  = lane >> 4;
    const __bf16* p = base + col * KST + hi * 8;
    v8bf lo8 = *(const v8bf*)(p);
    v8bf hi8 = *(const v8bf*)(p + 16);
    return __builtin_shufflevector(lo8, hi8, 0,1,2,3,4,5,6,7,8,9,10,11,12,13,14,15);
}

__global__ __launch_bounds__(NTHREADS)
void attn_fwd_kernel(const float* __restrict__ Q, const float* __restrict__ K,
                     const float* __restrict__ V, const unsigned char* __restrict__ mask,
                     float* __restrict__ O)
{
    // double-buffered raw fp32 staging for async copies
    __shared__ alignas(16) float  Kraw[2][BN * DDIM];   // 2 x 16 KB
    __shared__ alignas(16) float  Vraw[2][BN * DDIM];   // 2 x 16 KB
    __shared__ alignas(16) __bf16 Qs[BM * KST];         // Q tile, bf16 (m,d)
    __shared__ alignas(16) __bf16 Ks[BN * KST];         // K block, bf16 (n,d)
    __shared__ alignas(16) __bf16 Vt[DDIM * KST];       // V block, bf16 (d,n)
    __shared__ alignas(16) __bf16 Ps[BM * KST];         // P block, bf16 (m,n)
    __shared__ float Ss[BM * SST];                      // S scratch, fp32
    __shared__ float maskAdd[BN];
    __shared__ float rowM[BM], rowL[BM], rowA[BM];
    __shared__ float pmax[NTHREADS], psum[NTHREADS];

    const int t    = threadIdx.x;
    const int lane = t & 31;
    const int wave = t >> 5;
    const int hi   = lane >> 4;
    const int ln   = lane & 15;

    const int nQB = SEQ / BM;                  // 32
    const int qb  = blockIdx.x % nQB;
    const int h   = (blockIdx.x / nQB) % HEADS;
    const int b   = blockIdx.x / (nQB * HEADS);

    const long long bhBase = ((long long)(b * HEADS + h)) * SEQ * DDIM;
    const float* Qg = Q + bhBase + (long long)qb * BM * DDIM;
    const float* Kg = K + bhBase;
    const float* Vg = V + bhBase;
    const unsigned char* Mg = mask + (long long)b * SEQ;

    // ---- prologue: async-stage block 0 while we prepare Q --------------------
    #pragma unroll
    for (int i = 0; i < 8; ++i) {
        int e = i * NTHREADS + t;              // float4 index 0..1023
        async_copy_b128(lds_off(&Kraw[0][e * 4]), Kg + e * 4);
        async_copy_b128(lds_off(&Vraw[0][e * 4]), Vg + e * 4);
    }

    // Q tile -> bf16 LDS (synchronous; only once)
    #pragma unroll
    for (int i = 0; i < 8; ++i) {
        int e = i * NTHREADS + t;
        int r = e >> 4;
        int c = (e & 15) * 4;
        float4 f = *(const float4*)(Qg + r * DDIM + c);
        __bf16* dst = &Qs[r * KST + c];
        dst[0] = (__bf16)f.x; dst[1] = (__bf16)f.y;
        dst[2] = (__bf16)f.z; dst[3] = (__bf16)f.w;
    }
    if (t < BM) { rowM[t] = -INFINITY; rowL[t] = 0.0f; }
    __syncthreads();

    // Q A-fragments live in registers for the whole kernel (D=64 -> 2 K-chunks)
    v16bf qa0 = load_a_frag(&Qs[(wave * 16) * KST + 0],  lane);
    v16bf qa1 = load_a_frag(&Qs[(wave * 16) * KST + 32], lane);

    v8f o[4];
    #pragma unroll
    for (int i = 0; i < 4; ++i)
        #pragma unroll
        for (int r = 0; r < 8; ++r) o[i][r] = 0.0f;

    const float invScale = 0.125f;   // 1/sqrt(64)

    for (int kb = 0; kb < NBLK; ++kb) {
        const float* KrawB = Kraw[kb & 1];
        const float* VrawB = Vraw[kb & 1];

        wait_async0();        // this wave's copies for block kb have landed
        __syncthreads();      // ... and everyone else's; Ks/Vt/Ps free to reuse

        // issue async stage of block kb+1 into the other buffer
        if (kb + 1 < NBLK) {
            const float* Kn = Kg + (kb + 1) * BN * DDIM;
            const float* Vn = Vg + (kb + 1) * BN * DDIM;
            float* Kd = Kraw[(kb + 1) & 1];
            float* Vd = Vraw[(kb + 1) & 1];
            #pragma unroll
            for (int i = 0; i < 8; ++i) {
                int e = i * NTHREADS + t;
                async_copy_b128(lds_off(&Kd[e * 4]), Kn + e * 4);
                async_copy_b128(lds_off(&Vd[e * 4]), Vn + e * 4);
            }
        }

        // convert staged fp32 -> bf16: Ks row-major (n,d), Vt transposed (d,n)
        #pragma unroll
        for (int i = 0; i < 8; ++i) {
            int e = i * NTHREADS + t;
            int r = e >> 4;
            int c = (e & 15) * 4;
            float4 fk = *(const float4*)(KrawB + e * 4);
            __bf16* dk = &Ks[r * KST + c];
            dk[0] = (__bf16)fk.x; dk[1] = (__bf16)fk.y;
            dk[2] = (__bf16)fk.z; dk[3] = (__bf16)fk.w;
            float4 fv = *(const float4*)(VrawB + e * 4);
            Vt[(c + 0) * KST + r] = (__bf16)fv.x;
            Vt[(c + 1) * KST + r] = (__bf16)fv.y;
            Vt[(c + 2) * KST + r] = (__bf16)fv.z;
            Vt[(c + 3) * KST + r] = (__bf16)fv.w;
        }
        if (t < BN)
            maskAdd[t] = Mg[kb * BN + t] ? -INFINITY : 0.0f;
        __syncthreads();

        // ---- S = Q K^T : 4 n-tiles x 2 K-chunks of WMMA, then cap+mask ----
        #pragma unroll
        for (int tn = 0; tn < 4; ++tn) {
            v16bf kb0 = load_b_frag(&Ks[(tn * 16) * KST + 0],  lane);
            v16bf kb1 = load_b_frag(&Ks[(tn * 16) * KST + 32], lane);
            v8f acc;
            #pragma unroll
            for (int r = 0; r < 8; ++r) acc[r] = 0.0f;
            acc = __builtin_amdgcn_wmma_f32_16x16x32_bf16(false, qa0, false, kb0,
                                                          (short)0, acc, false, false);
            acc = __builtin_amdgcn_wmma_f32_16x16x32_bf16(false, qa1, false, kb1,
                                                          (short)0, acc, false, false);
            float madd = maskAdd[tn * 16 + ln];
            #pragma unroll
            for (int r = 0; r < 8; ++r) {
                float s = 10.0f * tanhf(acc[r] * invScale) + madd;
                int rowLoc = wave * 16 + r + hi * 8;
                Ss[rowLoc * SST + tn * 16 + ln] = s;
            }
        }
        __syncthreads();

        // ---- online softmax, all 128 lanes: 2 threads per row --------------
        {   // pass 1: partial max over half a row
            int row = t & 63, half = t >> 6;
            float rmax = -INFINITY;
            #pragma unroll 8
            for (int j = 0; j < 32; ++j)
                rmax = fmaxf(rmax, Ss[row * SST + half * 32 + j]);
            pmax[t] = rmax;
        }
        __syncthreads();
        if (t < BM) {   // pass 2: combine, update running max, alpha
            float mOld = rowM[t];
            float mNew = fmaxf(mOld, fmaxf(pmax[t], pmax[t + 64]));
            rowA[t] = (mNew == -INFINITY) ? 1.0f : __expf(mOld - mNew);
            rowM[t] = mNew;
        }
        __syncthreads();
        {   // pass 3: exponentiate + emit bf16 P, partial sums
            int row = t & 63, half = t >> 6;
            float mNew = rowM[row];
            float sum = 0.0f;
            if (mNew == -INFINITY) {
                #pragma unroll 8
                for (int j = 0; j < 32; ++j)
                    Ps[row * KST + half * 32 + j] = (__bf16)0.0f;
            } else {
                #pragma unroll 8
                for (int j = 0; j < 32; ++j) {
                    float e = __expf(Ss[row * SST + half * 32 + j] - mNew);
                    sum += e;
                    Ps[row * KST + half * 32 + j] = (__bf16)e;
                }
            }
            psum[t] = sum;
        }
        __syncthreads();
        if (t < BM)     // pass 4: running denominator
            rowL[t] = rowL[t] * rowA[t] + psum[t] + psum[t + 64];

        // ---- rescale O, then O += P @ V ------------------------------------
        #pragma unroll
        for (int r = 0; r < 8; ++r) {
            float a = rowA[wave * 16 + r + hi * 8];
            #pragma unroll
            for (int td = 0; td < 4; ++td) o[td][r] *= a;
        }
        v16bf pa0 = load_a_frag(&Ps[(wave * 16) * KST + 0],  lane);
        v16bf pa1 = load_a_frag(&Ps[(wave * 16) * KST + 32], lane);
        #pragma unroll
        for (int td = 0; td < 4; ++td) {
            v16bf vb0 = load_b_frag(&Vt[(td * 16) * KST + 0],  lane);
            v16bf vb1 = load_b_frag(&Vt[(td * 16) * KST + 32], lane);
            o[td] = __builtin_amdgcn_wmma_f32_16x16x32_bf16(false, pa0, false, vb0,
                                                            (short)0, o[td], false, false);
            o[td] = __builtin_amdgcn_wmma_f32_16x16x32_bf16(false, pa1, false, vb1,
                                                            (short)0, o[td], false, false);
        }
    }
    __syncthreads();

    // ---- normalize by row sums and store -----------------------------------
    #pragma unroll
    for (int r = 0; r < 8; ++r) {
        int rowLoc = wave * 16 + r + hi * 8;
        float inv = 1.0f / rowL[rowLoc];
        float* dst = O + bhBase + ((long long)qb * BM + rowLoc) * DDIM;
        #pragma unroll
        for (int td = 0; td < 4; ++td)
            dst[td * 16 + ln] = o[td][r] * inv;
    }
}

extern "C" void kernel_launch(void* const* d_in, const int* in_sizes, int n_in,
                              void* d_out, int out_size, void* d_ws, size_t ws_size,
                              hipStream_t stream) {
    const float* q = (const float*)d_in[0];
    const float* k = (const float*)d_in[1];
    const float* v = (const float*)d_in[2];
    const unsigned char* m = (const unsigned char*)d_in[3];
    float* out = (float*)d_out;
    dim3 grid(BATCH * HEADS * (SEQ / BM));
    attn_fwd_kernel<<<grid, NTHREADS, 0, stream>>>(q, k, v, m, out);
}
